// MTreeGRU_59141699666272
// MI455X (gfx1250) — matrix operand order repaired and checked by reading, not compiled, not run on hardware
//
#include <hip/hip_runtime.h>
#include <hip/hip_bf16.h>

// ---------------------------------------------------------------------------
// MTreeGRU on MI455X (gfx1250): bf16 WMMA implicit-GEMM 3x3 convs,
// shared-visual 1x1 GEMM + per-node rank-1 lang bias, device-side tree order.
// Round 3: software-pipelined conv loop (register prefetch of next K-chunk
// overlapping the 9-tap WMMA block), single LDS buffer (49.5 KB).
// ---------------------------------------------------------------------------

typedef __attribute__((ext_vector_type(16))) __bf16 v16bf;
typedef __attribute__((ext_vector_type(8)))  float  v8f;

union FragU {            // 16 bf16 values = 32 bytes
    uint4 q[2];
    v16bf v;
};

#define NODES 20
#define CH    512
#define IMW   64
#define HW    4096               // 64*64
#define CHW   (CH * HW)          // 2,097,152 floats = 8 MB
#define WPACK ((size_t)9 * CH * (2 * CH))   // packed bf16 weight elements
#define INUNITS (3 * 66 * 4)     // 792 8-channel input units per chunk
#define WUNITS  (9 * 64 * 4)     // 2304 uint4 weight units per chunk

__device__ __forceinline__ unsigned short f2bf(float f) {
    unsigned int u = __float_as_uint(f);
    unsigned int r = (u + 0x7FFFu + ((u >> 16) & 1u)) >> 16;   // RNE
    return (unsigned short)r;
}

// ---------------------------------------------------------------------------
// Tree metadata (single thread): tree[0..19]=topo order (children first, root
// last), tree[20..39]=childCount, tree[40..439]=childList[node][j].
// ---------------------------------------------------------------------------
__global__ void build_tree(const int* __restrict__ adj, int* __restrict__ tree) {
    if (threadIdx.x != 0 || blockIdx.x != 0) return;
    for (int i = 0; i < NODES; ++i) {
        int cc = 0;
        for (int j = 0; j < NODES; ++j)
            if (adj[i * NODES + j] > 0) tree[2 * NODES + i * NODES + cc++] = j;
        tree[NODES + i] = cc;
    }
    unsigned emitted = 0;
    int k = 0;
    for (int pass = 0; pass < NODES && k < NODES; ++pass) {
        for (int i = 0; i < NODES; ++i) {
            if (emitted & (1u << i)) continue;
            int cc = tree[NODES + i];
            bool ok = true;
            for (int j = 0; j < cc; ++j)
                if (!(emitted & (1u << tree[2 * NODES + i * NODES + j]))) { ok = false; break; }
            if (ok) { tree[k++] = i; emitted |= (1u << i); }
        }
    }
}

// ---------------------------------------------------------------------------
// Pack conv weights: fp32 [oc][ic][tap] -> bf16 [tap][oc][ic]  (done once).
// ---------------------------------------------------------------------------
__global__ void pack_weights(const float* __restrict__ w,
                             unsigned short* __restrict__ wp) {
    size_t i = (size_t)blockIdx.x * blockDim.x + threadIdx.x;  // over [tap][oc][ic]
    if (i >= WPACK) return;
    int tap = (int)(i >> 19);                 // 512*1024 = 2^19
    int rem = (int)(i & 0x7FFFF);
    int oc  = rem >> 10;
    int ic  = rem & 1023;
    wp[i] = f2bf(w[((size_t)oc * (2 * CH) + ic) * 9 + tap]);
}

// ---------------------------------------------------------------------------
// xlang[node][oc] = mconv_b[oc] + sum_cl lang[node][cl] * mconv_w[oc][512+cl]
// ---------------------------------------------------------------------------
__global__ void xlang_kernel(const float* __restrict__ lang,
                             const float* __restrict__ mw,
                             const float* __restrict__ mb,
                             float* __restrict__ xlang) {
    int i = blockIdx.x * blockDim.x + threadIdx.x;
    if (i >= NODES * CH) return;
    int node = i >> 9;
    int oc   = i & (CH - 1);
    float s = mb[oc];
    const float* l = lang + node * 300;
    const float* w = mw + (size_t)oc * 812 + 512;
    for (int cl = 0; cl < 300; ++cl) s += l[cl] * w[cl];
    xlang[i] = s;
}

// ---------------------------------------------------------------------------
// xvis[oc][p] = sum_cv vis[cv][p] * mconv_w[oc][cv]   (M=4096, N=512, K=512)
// ---------------------------------------------------------------------------
__global__ __launch_bounds__(256) void mconv_vis_wmma(const float* __restrict__ vis,
                                                      const float* __restrict__ mw,
                                                      float* __restrict__ xvis) {
    __shared__ __align__(16) unsigned short sA[64 * 32];   // [pixel][ic]
    __shared__ __align__(16) unsigned short sB[64 * 32];   // [oc][ic]

    const int tid   = threadIdx.x;
    const int lane  = tid & 31;
    const int wid   = tid >> 5;
    const int waveM = wid & 3;
    const int waveN = wid >> 2;
    const int pixBase = blockIdx.x * 64;
    const int ocBase  = blockIdx.y * 64;

    v8f acc0 = {}; v8f acc1 = {};

    for (int ic0 = 0; ic0 < CH; ic0 += 32) {
        {   // sA: one 8-channel unit per thread
            int c8 = tid & 3;
            int px = tid >> 2;
            unsigned short t8[8];
            const float* s = &vis[(size_t)(ic0 + c8 * 8) * HW + pixBase + px];
#pragma unroll
            for (int k = 0; k < 8; ++k) t8[k] = f2bf(s[(size_t)k * HW]);
            *(uint4*)&sA[(px << 5) + c8 * 8] = *(const uint4*)t8;
        }
        {   // sB: one 8-channel unit per thread (contiguous fp32 source)
            int c8 = tid & 3;
            int oc = tid >> 2;
            unsigned short t8[8];
            const float* s = &mw[(size_t)(ocBase + oc) * 812 + ic0 + c8 * 8];
#pragma unroll
            for (int k = 0; k < 8; ++k) t8[k] = f2bf(s[k]);
            *(uint4*)&sB[(oc << 5) + c8 * 8] = *(const uint4*)t8;
        }
        __syncthreads();

        const int mrow  = lane & 15;
        const int khalf = (lane >> 4) << 3;
        const int kbase = (lane >> 4) << 4;
        const uint4* ap = (const uint4*)&sA[(waveM * 16 + mrow) * 32 + kbase];
        FragU a; a.q[0] = ap[0]; a.q[1] = ap[1];

        const uint4* b0p = (const uint4*)&sB[(waveN * 32 + (lane & 15)) * 32 + kbase];
        const uint4* b1p = (const uint4*)&sB[(waveN * 32 + 16 + (lane & 15)) * 32 + kbase];
        FragU b0; b0.q[0] = b0p[0]; b0.q[1] = b0p[1];
        FragU b1; b1.q[0] = b1p[0]; b1.q[1] = b1p[1];

        acc0 = __builtin_amdgcn_wmma_f32_16x16x32_bf16(false, a.v, false, b0.v,
                                                       (short)0, acc0, false, false);
        acc1 = __builtin_amdgcn_wmma_f32_16x16x32_bf16(false, a.v, false, b1.v,
                                                       (short)0, acc1, false, false);
        __syncthreads();
    }

    const int n   = lane & 15;
    const int mhi = (lane >> 4) << 3;
    for (int t = 0; t < 2; ++t) {
        int ocg = ocBase + waveN * 32 + t * 16 + n;
        v8f av = t ? acc1 : acc0;
#pragma unroll
        for (int r = 0; r < 8; ++r) {
            int pix = pixBase + waveM * 16 + r + mhi;
            xvis[(size_t)ocg * HW + pix] = av[r];
        }
    }
}

// ---------------------------------------------------------------------------
// ch_sum = sum_children h;  reset_hidden = 0
// ---------------------------------------------------------------------------
__global__ void accum_children(const float* __restrict__ hbuf,
                               float* __restrict__ chsum,
                               float* __restrict__ rhid,
                               const int* __restrict__ tree, int step) {
    int i = blockIdx.x * blockDim.x + threadIdx.x;
    if (i >= CHW) return;
    int node = tree[step];
    int cc   = tree[NODES + node];
    float s = 0.f;
    for (int j = 0; j < cc; ++j) {
        int c = tree[2 * NODES + node * NODES + j];
        s += hbuf[(size_t)c * CHW + i];
    }
    chsum[i] = s;
    rhid[i]  = 0.f;
}

// ---------------------------------------------------------------------------
// 3x3 conv (1024 -> 512 ch) as software-pipelined bf16 WMMA implicit GEMM.
// Input channels 0..511  = xvis + xlang[node] (rank-1 bias), zero halo;
//       channels 512..1023 = inB (h_child / ch_sum / reset_hidden).
// wp: packed bf16 weights [tap][oc][ic].
// mode 0: reset  -> atomicAdd(dst, sigmoid(v) * h_child)   (grid.z = child)
// mode 1: update -> dst = sigmoid(v)
// mode 2: output -> dst = tanh(v)
// ---------------------------------------------------------------------------
__global__ __launch_bounds__(256) void conv3x3_wmma(
        const float* __restrict__ xvis, const float* __restrict__ xlangAll,
        const float* __restrict__ hbuf, const float* __restrict__ inBdirect,
        const unsigned short* __restrict__ wp, const float* __restrict__ bias,
        float* __restrict__ dst, const int* __restrict__ tree,
        int step, int mode) {
    __shared__ __align__(16) unsigned short sIn[3 * 66 * 32];   // [row][col][ic]
    __shared__ __align__(16) unsigned short sW[9 * 64 * 32];    // [tap][oc][ic]

    const int node = tree[step];
    const float* inB;
    const float* mult = nullptr;
    if (mode == 0) {
        int cc = tree[NODES + node];
        if ((int)blockIdx.z >= cc) return;
        int child = tree[2 * NODES + node * NODES + (int)blockIdx.z];
        inB  = hbuf + (size_t)child * CHW;
        mult = inB;
    } else {
        inB = inBdirect;
    }
    const float* abias = xlangAll + node * CH;

    const int tid   = threadIdx.x;
    const int lane  = tid & 31;
    const int wid   = tid >> 5;
    const int waveM = wid & 3;
    const int waveN = wid >> 2;
    const int tileY  = blockIdx.x;        // image row (64 pixels)
    const int ocBase = blockIdx.y * 64;

    // --- per-thread unit geometry (chunk-invariant) ---
    int ucol[4], urow[4];
    bool uok[4];
    const int c8 = tid & 3;               // same for all 4 units (256 % 4 == 0)
#pragma unroll
    for (int un = 0; un < 4; ++un) {
        int u   = tid + un * 256;
        int rc  = u >> 2;
        ucol[un] = rc % 66;
        urow[un] = rc / 66;
        int gx = ucol[un] - 1;
        int gy = tileY + urow[un] - 1;
        uok[un] = (u < INUNITS) && (gx >= 0 && gx < IMW && gy >= 0 && gy < IMW);
    }

    v8f acc0 = {}; v8f acc1 = {};
    float inReg[4][8];
    float abReg[8];
    uint4 wReg[9];

    // ---- prologue: issue loads for chunk 0 ----
    {
        const int ic0 = 0;
        const float* src = xvis;          // chunk 0 is the x half
#pragma unroll
        for (int un = 0; un < 4; ++un) {
            if (uok[un]) {
                const float* p = src + (size_t)(c8 * 8) * HW
                               + (tileY + urow[un] - 1) * IMW + (ucol[un] - 1);
#pragma unroll
                for (int k = 0; k < 8; ++k) inReg[un][k] = p[(size_t)k * HW];
            }
        }
#pragma unroll
        for (int k = 0; k < 8; ++k) abReg[k] = abias[ic0 + c8 * 8 + k];
#pragma unroll
        for (int i = 0; i < 9; ++i) {
            int idx = tid + i * 256;
            int wc8 = idx & 3;
            int oc  = (idx >> 2) & 63;
            int tap = idx >> 8;
            wReg[i] = *(const uint4*)
                &wp[((size_t)tap * CH + ocBase + oc) * (2 * CH) + ic0 + wc8 * 8];
        }
    }

    for (int ic0 = 0; ic0 < 2 * CH; ic0 += 32) {
        const bool isX = (ic0 < CH);
        // ---- drain registers (chunk ic0) into LDS ----
#pragma unroll
        for (int un = 0; un < 4; ++un) {
            int u = tid + un * 256;
            if (u < INUNITS) {
                unsigned short t8[8];
                if (uok[un]) {
#pragma unroll
                    for (int k = 0; k < 8; ++k) {
                        float v = inReg[un][k];
                        if (isX) v += abReg[k];
                        t8[k] = f2bf(v);
                    }
                } else {
#pragma unroll
                    for (int k = 0; k < 8; ++k) t8[k] = 0;
                }
                *(uint4*)&sIn[((urow[un] * 66 + ucol[un]) << 5) + c8 * 8] =
                    *(const uint4*)t8;
            }
        }
#pragma unroll
        for (int i = 0; i < 9; ++i) ((uint4*)sW)[tid + i * 256] = wReg[i];

        // ---- issue loads for chunk ic0+32 (fly during compute) ----
        const int icn = ic0 + 32;
        if (icn < 2 * CH) {
            const bool nX = (icn < CH);
            const float* src = nX ? (xvis + (size_t)icn * HW)
                                  : (inB + (size_t)(icn - CH) * HW);
#pragma unroll
            for (int un = 0; un < 4; ++un) {
                if (uok[un]) {
                    const float* p = src + (size_t)(c8 * 8) * HW
                                   + (tileY + urow[un] - 1) * IMW + (ucol[un] - 1);
#pragma unroll
                    for (int k = 0; k < 8; ++k) inReg[un][k] = p[(size_t)k * HW];
                }
            }
#pragma unroll
            for (int k = 0; k < 8; ++k)
                abReg[k] = nX ? abias[icn + c8 * 8 + k] : 0.f;
#pragma unroll
            for (int i = 0; i < 9; ++i) {
                int idx = tid + i * 256;
                int wc8 = idx & 3;
                int oc  = (idx >> 2) & 63;
                int tap = idx >> 8;
                wReg[i] = *(const uint4*)
                    &wp[((size_t)tap * CH + ocBase + oc) * (2 * CH) + icn + wc8 * 8];
            }
        }
        __syncthreads();

        // ---- 9-tap WMMA block on chunk ic0 (prefetch loads in flight) ----
        const int mrow  = lane & 15;
        const int khalf = (lane >> 4) << 3;
        const int kbase = (lane >> 4) << 4;
        const int n16   = lane & 15;

        for (int tap = 0; tap < 9; ++tap) {
            int ky = tap / 3, kx = tap - ky * 3;
            int col = waveM * 16 + mrow + kx;      // 0..65
            const uint4* ap = (const uint4*)&sIn[((ky * 66 + col) << 5) + kbase];
            FragU a; a.q[0] = ap[0]; a.q[1] = ap[1];

            const uint4* b0p = (const uint4*)&sW[((tap * 64 + waveN * 32 + n16) << 5) + kbase];
            const uint4* b1p = (const uint4*)&sW[((tap * 64 + waveN * 32 + 16 + n16) << 5) + kbase];
            FragU b0; b0.q[0] = b0p[0]; b0.q[1] = b0p[1];
            FragU b1; b1.q[0] = b1p[0]; b1.q[1] = b1p[1];

            acc0 = __builtin_amdgcn_wmma_f32_16x16x32_bf16(false, a.v, false, b0.v,
                                                           (short)0, acc0, false, false);
            acc1 = __builtin_amdgcn_wmma_f32_16x16x32_bf16(false, a.v, false, b1.v,
                                                           (short)0, acc1, false, false);
        }
        __syncthreads();   // all waves done reading before next drain
    }

    // ---- epilogue ----
    const int n   = lane & 15;
    const int mhi = (lane >> 4) << 3;
    for (int t = 0; t < 2; ++t) {
        int ocg = ocBase + waveN * 32 + t * 16 + n;
        float bv = bias[ocg];
        v8f av = t ? acc1 : acc0;
#pragma unroll
        for (int r = 0; r < 8; ++r) {
            int pix = tileY * IMW + waveM * 16 + r + mhi;
            size_t off = (size_t)ocg * HW + pix;
            float v = av[r] + bv;
            if (mode == 0) {
                float rv = 1.f / (1.f + __expf(-v));
                atomicAdd(&dst[off], rv * mult[off]);
            } else if (mode == 1) {
                dst[off] = 1.f / (1.f + __expf(-v));
            } else {
                dst[off] = tanhf(v);
            }
        }
    }
}

// ---------------------------------------------------------------------------
// h[node] = (1-z)*o + z*ch_sum
// ---------------------------------------------------------------------------
__global__ void combine_kernel(float* __restrict__ hbuf,
                               const float* __restrict__ zbuf,
                               const float* __restrict__ obuf,
                               const float* __restrict__ chsum,
                               const int* __restrict__ tree, int step) {
    int i = blockIdx.x * blockDim.x + threadIdx.x;
    if (i >= CHW) return;
    int node = tree[step];
    float z = zbuf[i];
    hbuf[(size_t)node * CHW + i] = (1.f - z) * obuf[i] + z * chsum[i];
}

__global__ void copy_root(const float* __restrict__ hbuf,
                          const int* __restrict__ tree,
                          float* __restrict__ out) {
    int i = blockIdx.x * blockDim.x + threadIdx.x;
    if (i >= CHW) return;
    out[i] = hbuf[(size_t)tree[NODES - 1] * CHW + i];
}

// ---------------------------------------------------------------------------
extern "C" void kernel_launch(void* const* d_in, const int* in_sizes, int n_in,
                              void* d_out, int out_size, void* d_ws, size_t ws_size,
                              hipStream_t stream) {
    const float* vis   = (const float*)d_in[0];
    const float* lang  = (const float*)d_in[1];
    const int*   adj   = (const int*)d_in[2];
    const float* mw    = (const float*)d_in[3];
    const float* mb    = (const float*)d_in[4];
    const float* rw    = (const float*)d_in[5];
    const float* rb    = (const float*)d_in[6];
    const float* uw    = (const float*)d_in[7];
    const float* ub    = (const float*)d_in[8];
    const float* ow    = (const float*)d_in[9];
    const float* ob    = (const float*)d_in[10];
    float* out = (float*)d_out;

    // workspace layout (floats first, all 16B aligned by construction)
    float* xvis  = (float*)d_ws;                     // CHW
    float* xlang = xvis + (size_t)CHW;               // NODES*CH
    float* hbuf  = xlang + NODES * CH;               // NODES*CHW
    float* chsum = hbuf + (size_t)NODES * CHW;       // CHW
    float* rhid  = chsum + (size_t)CHW;              // CHW
    float* zbuf  = rhid + (size_t)CHW;               // CHW
    float* obuf  = zbuf + (size_t)CHW;               // CHW
    unsigned short* wpR = (unsigned short*)(obuf + (size_t)CHW);
    unsigned short* wpU = wpR + WPACK;
    unsigned short* wpO = wpU + WPACK;
    int*   tree  = (int*)(wpO + WPACK);              // 440 ints

    build_tree<<<1, 1, 0, stream>>>(adj, tree);
    const int pb = (int)((WPACK + 255) / 256);
    pack_weights<<<pb, 256, 0, stream>>>(rw, wpR);
    pack_weights<<<pb, 256, 0, stream>>>(uw, wpU);
    pack_weights<<<pb, 256, 0, stream>>>(ow, wpO);
    xlang_kernel<<<(NODES * CH + 255) / 256, 256, 0, stream>>>(lang, mw, mb, xlang);
    mconv_vis_wmma<<<dim3(64, 8), 256, 0, stream>>>(vis, mw, xvis);

    const int eb = CHW / 256;   // elementwise blocks
    for (int step = 0; step < NODES; ++step) {
        accum_children<<<eb, 256, 0, stream>>>(hbuf, chsum, rhid, tree, step);
        // reset: one conv per child edge, r*h accumulated atomically
        conv3x3_wmma<<<dim3(64, 8, NODES - 1), 256, 0, stream>>>(
            xvis, xlang, hbuf, nullptr, wpR, rb, rhid, tree, step, 0);
        // update gate
        conv3x3_wmma<<<dim3(64, 8, 1), 256, 0, stream>>>(
            xvis, xlang, hbuf, chsum, wpU, ub, zbuf, tree, step, 1);
        // candidate (tanh) on [x ; reset_hidden]
        conv3x3_wmma<<<dim3(64, 8, 1), 256, 0, stream>>>(
            xvis, xlang, hbuf, rhid, wpO, ob, obuf, tree, step, 2);
        combine_kernel<<<eb, 256, 0, stream>>>(hbuf, zbuf, obuf, chsum, tree, step);
    }
    copy_root<<<eb, 256, 0, stream>>>(hbuf, tree, out);
}
